// MultiHeadAttention_81157702025556
// MI455X (gfx1250) — matrix operand order repaired and checked by reading, not compiled, not run on hardware
//
#include <hip/hip_runtime.h>
#include <hip/hip_bf16.h>

// Problem constants (match reference)
#define BB 4
#define SS 2048
#define DD 1024
#define HH 16
#define HDIM 64

typedef __attribute__((ext_vector_type(16))) __bf16 v16bf;
typedef __attribute__((ext_vector_type(8)))  __bf16 v8bf;
typedef __attribute__((ext_vector_type(8)))  float  v8f;
typedef __attribute__((ext_vector_type(4)))  unsigned int v4u;
typedef __attribute__((ext_vector_type(8)))  int v8i;
typedef __attribute__((ext_vector_type(4)))  int v4i;

#if defined(__has_builtin)
#if __has_builtin(__builtin_amdgcn_tensor_load_to_lds) && \
    __has_builtin(__builtin_amdgcn_s_wait_tensorcnt)
#define HAVE_TDM 1
#endif
#endif
#ifndef HAVE_TDM
#define HAVE_TDM 0
#endif

#define WMMA_BF16(a, b, c) \
  __builtin_amdgcn_wmma_f32_16x16x32_bf16(false, (a), false, (b), (short)0, (c), false, false)

// ---- WMMA operand loaders (A/B: this code's contiguous K per lane half, same order in both; C/D: ISA layout) ----
// A (16x32 bf16): lane l -> row l&15, K = kb..kb+15, kb=(l<16)?0:16.
// B (32x16 bf16): lane l -> col l&15, K = kb..kb+15, kb=(l<16)?0:16.
// C/D (16x16 f32): VGPR i -> row i + 8*(l>=16), col l&15.

__device__ __forceinline__ v16bf load_a_f32(const float* __restrict__ rowk0, int lane) {
  const int kb = (lane < 16) ? 0 : 16;
  const float4 x0 = *(const float4*)(rowk0 + kb);
  const float4 x1 = *(const float4*)(rowk0 + kb + 4);
  const float4 x2 = *(const float4*)(rowk0 + kb + 8);
  const float4 x3 = *(const float4*)(rowk0 + kb + 12);
  v16bf r;
  r[0]=(__bf16)x0.x; r[1]=(__bf16)x0.y; r[2]=(__bf16)x0.z; r[3]=(__bf16)x0.w;
  r[4]=(__bf16)x1.x; r[5]=(__bf16)x1.y; r[6]=(__bf16)x1.z; r[7]=(__bf16)x1.w;
  r[8]=(__bf16)x2.x; r[9]=(__bf16)x2.y; r[10]=(__bf16)x2.z; r[11]=(__bf16)x2.w;
  r[12]=(__bf16)x3.x; r[13]=(__bf16)x3.y; r[14]=(__bf16)x3.z; r[15]=(__bf16)x3.w;
  return r;
}

__device__ __forceinline__ v16bf load_b_f32(const float* __restrict__ colk0, int lane) {
  const int kb = (lane < 16) ? 0 : 16;
  const float4 x0 = *(const float4*)(colk0 + kb);
  const float4 x1 = *(const float4*)(colk0 + kb + 4);
  const float4 x2 = *(const float4*)(colk0 + kb + 8);
  const float4 x3 = *(const float4*)(colk0 + kb + 12);
  v16bf r;
  r[0]=(__bf16)x0.x; r[1]=(__bf16)x0.y; r[2]=(__bf16)x0.z; r[3]=(__bf16)x0.w;
  r[4]=(__bf16)x1.x; r[5]=(__bf16)x1.y; r[6]=(__bf16)x1.z; r[7]=(__bf16)x1.w;
  r[8]=(__bf16)x2.x; r[9]=(__bf16)x2.y; r[10]=(__bf16)x2.z; r[11]=(__bf16)x2.w;
  r[12]=(__bf16)x3.x; r[13]=(__bf16)x3.y; r[14]=(__bf16)x3.z; r[15]=(__bf16)x3.w;
  return r;
}

__device__ __forceinline__ v16bf load_a_bf16(const __bf16* __restrict__ rowk0, int lane) {
  const int kb = (lane < 16) ? 0 : 16;
  const v8bf lo = *(const v8bf*)(rowk0 + kb);
  const v8bf hi = *(const v8bf*)(rowk0 + kb + 8);
  v16bf r;
#pragma unroll
  for (int i = 0; i < 8; ++i) { r[i] = lo[i]; r[8 + i] = hi[i]; }
  return r;
}

__device__ __forceinline__ v16bf load_b_bf16(const __bf16* __restrict__ colk0, int lane) {
  const int kb = (lane < 16) ? 0 : 16;
  return *(const v16bf*)(colk0 + kb);
}

// =====================================================================
// GEMM: Y[m,n] = sum_k X[m,k]*W[n,k] + bias[n]   (X @ W^T + b)
// Block = 4 waves = 64(M) x 64(N) tile; wave w -> 64x16 (4 accumulators).
// A-panel (64 x 32 f32) staged to LDS via TDM (tensor_load_to_lds) with
// LDS padding: after every 32 DWORDs insert 4 DWORDs -> 36-float row stride,
// conflict-free for 16 consecutive rows (36k mod 64 all distinct).
// Double-buffered: s_wait_tensorcnt + barriers. Fallback: float4 coop copy.
// =====================================================================
#define AROWS 64
#define ASTRIDE 36

template <bool OUT_BF16, bool VTRANS>
__global__ __launch_bounds__(128) void gemm_xwt_kernel(
    const float* __restrict__ X, const float* __restrict__ W,
    const float* __restrict__ bias, void* __restrict__ out,
    int M, int N, int K) {
  __shared__ float sa[2][AROWS][ASTRIDE];
  const int wid = threadIdx.x >> 5;
  const int lane = threadIdx.x & 31;
  const int ntb = N >> 6;
  const int mt = blockIdx.x / ntb;
  const int nt = blockIdx.x - mt * ntb;

  const int col = nt * 64 + wid * 16 + (lane & 15);
  const float* bcol = W + (size_t)col * K;
  const float* atile = X + (size_t)(mt * 64) * K;

  v8f c[4] = {};

  auto stage = [&](int buf, int k0) {
#if HAVE_TDM
    if (wid == 0) {
      const uint64_t ga = (uint64_t)(uintptr_t)(atile + k0);
      const uint32_t lds_off = (uint32_t)(uintptr_t)(&sa[buf][0][0]);
      // D# group0: count=1 | lds_addr | global_addr lo | global_addr hi + type=2
      v4u g0 = { 1u, lds_off, (uint32_t)ga,
                 (uint32_t)(ga >> 32) | (2u << 30) };
      // D# group1: data_size=4B, pad_enable, pad_interval=32 DW, pad_amount=4 DW;
      // tensor_dim0=K, tensor_dim1=M, tile_dim0=32, tile_dim1=64, dim0_stride=K.
      v8i g1 = { (int)0x07120000u,
                 (int)((uint32_t)K << 16),
                 (int)(((uint32_t)K >> 16) | ((uint32_t)M << 16)),
                 (int)((((uint32_t)M >> 16) & 0xFFFFu) | (32u << 16)),
                 (int)64,
                 (int)(uint32_t)K,
                 0, 0 };
      v4i z4 = { 0, 0, 0, 0 };
      v8i z8 = { 0, 0, 0, 0, 0, 0, 0, 0 };
      // 6-arg toolchain variant: (g0, g1, g2, g3, g4, cpol)
      __builtin_amdgcn_tensor_load_to_lds(g0, g1, z4, z4, z8, 0);
    }
#else
    const int t = threadIdx.x;
    const int r = t >> 1;               // 0..63
    const int cc = (t & 1) * 16;        // 0 or 16
    const float* src = atile + (size_t)r * K + k0 + cc;
#pragma unroll
    for (int j = 0; j < 4; ++j)
      *(float4*)&sa[buf][r][cc + j * 4] = *(const float4*)(src + j * 4);
#endif
  };

  const int NK = K >> 5;
  stage(0, 0);
  for (int kt = 0; kt < NK; ++kt) {
    const int buf = kt & 1;
#if HAVE_TDM
    if (wid == 0) __builtin_amdgcn_s_wait_tensorcnt(0);
#endif
    __syncthreads();
    if (kt + 1 < NK) stage(buf ^ 1, (kt + 1) * 32);

    const v16bf b = load_b_f32(bcol + kt * 32, lane);
#pragma unroll
    for (int m = 0; m < 4; ++m) {
      const v16bf a = load_a_f32(&sa[buf][m * 16 + (lane & 15)][0], lane);
      c[m] = WMMA_BF16(a, b, c[m]);
    }
    __syncthreads();
  }

  const float bn = bias[col];
  const int hi = (lane >> 4) << 3;
#pragma unroll
  for (int m = 0; m < 4; ++m) {
#pragma unroll
    for (int i = 0; i < 8; ++i) {
      const int r = mt * 64 + m * 16 + hi + i;
      const float v = c[m][i] + bn;
      if (OUT_BF16) {
        __bf16* o = (__bf16*)out;
        if (VTRANS) {
          const int b_ = r / SS, s_ = r - b_ * SS;
          o[(size_t)b_ * DD * SS + (size_t)col * SS + s_] = (__bf16)v;
        } else {
          o[(size_t)r * N + col] = (__bf16)v;
        }
      } else {
        ((float*)out)[(size_t)r * N + col] = v;
      }
    }
  }
}

// =====================================================================
// Scores: S[b,h,q,k] = (Q.K)/8 -> attn region (pre-softmax).
// One wave owns a (b,h,16-query) strip; Q fragments stay in registers and
// are reused across all 128 key tiles (2 WMMAs per tile over HD=64).
// =====================================================================
__global__ __launch_bounds__(128) void attn_scores_kernel(
    const __bf16* __restrict__ Qbf, const __bf16* __restrict__ Kbf,
    float* __restrict__ attn) {
  const int wid = threadIdx.x >> 5;
  const int lane = threadIdx.x & 31;
  const int id = blockIdx.x * 4 + wid;   // b*2^11 + h*2^7 + qt
  const int qt = id & 127;
  const int h = (id >> 7) & 15;
  const int b = id >> 11;

  const __bf16* qp = Qbf + (size_t)(b * SS + qt * 16 + (lane & 15)) * DD + h * HDIM;
  const v16bf a0 = load_a_bf16(qp, lane);
  const v16bf a1 = load_a_bf16(qp + 32, lane);

  const __bf16* kbase = Kbf + (size_t)(b * SS + (lane & 15)) * DD + h * HDIM;
  float* orow = attn + ((size_t)(b * HH + h) * SS + qt * 16) * SS;
  const int hi = (lane >> 4) << 3;

  for (int kt = 0; kt < 128; ++kt) {
    const __bf16* kp = kbase + (size_t)kt * 16 * DD;
    __builtin_prefetch(kp + 16 * DD, 0, 1);
    v8f c = {};
    c = WMMA_BF16(a0, load_b_bf16(kp, lane), c);
    c = WMMA_BF16(a1, load_b_bf16(kp + 32, lane), c);
#pragma unroll
    for (int i = 0; i < 8; ++i)
      orow[(size_t)(hi + i) * SS + kt * 16 + (lane & 15)] = c[i] * 0.125f;
  }
}

// ---- In-place row softmax over attn region (rows of 2048) ----
__global__ __launch_bounds__(256) void softmax_rows_kernel(float* __restrict__ attn) {
  __shared__ float red[256];
  float* row = attn + (size_t)blockIdx.x * SS;
  const int t = threadIdx.x;

  float4 x0 = ((const float4*)row)[t];
  float4 x1 = ((const float4*)row)[t + 256];
  float m = fmaxf(fmaxf(fmaxf(x0.x, x0.y), fmaxf(x0.z, x0.w)),
                  fmaxf(fmaxf(x1.x, x1.y), fmaxf(x1.z, x1.w)));
  red[t] = m;
  __syncthreads();
  for (int s = 128; s > 0; s >>= 1) {
    if (t < s) red[t] = fmaxf(red[t], red[t + s]);
    __syncthreads();
  }
  m = red[0];
  __syncthreads();

  float4 e0, e1;
  e0.x = __expf(x0.x - m); e0.y = __expf(x0.y - m);
  e0.z = __expf(x0.z - m); e0.w = __expf(x0.w - m);
  e1.x = __expf(x1.x - m); e1.y = __expf(x1.y - m);
  e1.z = __expf(x1.z - m); e1.w = __expf(x1.w - m);
  red[t] = (e0.x + e0.y + e0.z + e0.w) + (e1.x + e1.y + e1.z + e1.w);
  __syncthreads();
  for (int s = 128; s > 0; s >>= 1) {
    if (t < s) red[t] += red[t + s];
    __syncthreads();
  }
  const float inv = 1.0f / red[0];

  e0.x *= inv; e0.y *= inv; e0.z *= inv; e0.w *= inv;
  e1.x *= inv; e1.y *= inv; e1.z *= inv; e1.w *= inv;
  ((float4*)row)[t] = e0;
  ((float4*)row)[t + 256] = e1;
}

// =====================================================================
// Context: ctx[b,q,h*64+n] = sum_k P[b,h,q,k] * V[b,k,h*64+n].
// Register-blocked 32(q) x 64(n) per wave: 8 accumulators; A fragments
// (attn f32 -> bf16 on the fly) reused x4, B fragments (Vt bf16, K-contiguous
// columns) reused x2. K loop over S=2048.
// =====================================================================
__global__ __launch_bounds__(128) void attn_context_kernel(
    const float* __restrict__ attn, const __bf16* __restrict__ Vt,
    float* __restrict__ ctx) {
  const int wid = threadIdx.x >> 5;
  const int lane = threadIdx.x & 31;
  const int id = blockIdx.x * 4 + wid;   // b*2^10 + h*2^6 + qt
  const int qt = id & 63;
  const int h = (id >> 6) & 15;
  const int b = id >> 10;

  const float* arow0 =
      attn + ((size_t)(b * HH + h) * SS + qt * 32 + (lane & 15)) * SS;
  const float* arow1 = arow0 + (size_t)16 * SS;
  const __bf16* vbase = Vt + (size_t)b * DD * SS + (size_t)(h * HDIM + (lane & 15)) * SS;

  v8f c[2][4] = {};
  for (int k0 = 0; k0 < SS; k0 += 32) {
    __builtin_prefetch(arow0 + k0 + 128, 0, 1);
    __builtin_prefetch(arow1 + k0 + 128, 0, 1);
    const v16bf a0 = load_a_f32(arow0 + k0, lane);
    const v16bf a1 = load_a_f32(arow1 + k0, lane);
#pragma unroll
    for (int j = 0; j < 4; ++j) {
      const v16bf bb = load_b_bf16(vbase + (size_t)j * 16 * SS + k0, lane);
      c[0][j] = WMMA_BF16(a0, bb, c[0][j]);
      c[1][j] = WMMA_BF16(a1, bb, c[1][j]);
    }
  }

  const int hi = (lane >> 4) << 3;
#pragma unroll
  for (int m = 0; m < 2; ++m)
#pragma unroll
    for (int j = 0; j < 4; ++j)
#pragma unroll
      for (int i = 0; i < 8; ++i)
        ctx[(size_t)(b * SS + qt * 32 + m * 16 + hi + i) * DD +
            h * HDIM + j * 16 + (lane & 15)] = c[m][j][i];
}

// ---- out = LayerNorm(oproj + query) * gamma + beta ----
__global__ __launch_bounds__(256) void residual_ln_kernel(
    const float* __restrict__ oproj, const float* __restrict__ query,
    const float* __restrict__ gamma, const float* __restrict__ beta,
    float* __restrict__ out) {
  __shared__ float red[256];
  const int t = threadIdx.x;
  const size_t base = (size_t)blockIdx.x * DD;

  float4 x = ((const float4*)(oproj + base))[t];
  const float4 q = ((const float4*)(query + base))[t];
  x.x += q.x; x.y += q.y; x.z += q.z; x.w += q.w;

  red[t] = x.x + x.y + x.z + x.w;
  __syncthreads();
  for (int s = 128; s > 0; s >>= 1) {
    if (t < s) red[t] += red[t + s];
    __syncthreads();
  }
  const float mu = red[0] * (1.0f / DD);
  __syncthreads();

  const float dx = x.x - mu, dy = x.y - mu, dz = x.z - mu, dw = x.w - mu;
  red[t] = dx * dx + dy * dy + dz * dz + dw * dw;
  __syncthreads();
  for (int s = 128; s > 0; s >>= 1) {
    if (t < s) red[t] += red[t + s];
    __syncthreads();
  }
  const float inv = rsqrtf(red[0] * (1.0f / DD) + 1e-5f);

  const float4 g = ((const float4*)gamma)[t];
  const float4 bt = ((const float4*)beta)[t];
  float4 o;
  o.x = dx * inv * g.x + bt.x;
  o.y = dy * inv * g.y + bt.y;
  o.z = dz * inv * g.z + bt.z;
  o.w = dw * inv * g.w + bt.w;
  ((float4*)(out + base))[t] = o;
}

extern "C" void kernel_launch(void* const* d_in, const int* in_sizes, int n_in,
                              void* d_out, int out_size, void* d_ws, size_t ws_size,
                              hipStream_t stream) {
  const float* query = (const float*)d_in[0];
  const float* key_t = (const float*)d_in[1];
  const float* value = (const float*)d_in[2];
  const float* Wq = (const float*)d_in[3];
  const float* bq = (const float*)d_in[4];
  const float* Wk = (const float*)d_in[5];
  const float* bk = (const float*)d_in[6];
  const float* Wv = (const float*)d_in[7];
  const float* bv = (const float*)d_in[8];
  const float* Wo = (const float*)d_in[9];
  const float* bo = (const float*)d_in[10];
  const float* gamma = (const float*)d_in[11];
  const float* beta = (const float*)d_in[12];

  float* out_p = (float*)d_out;                           // [B,S,D]
  float* attn_p = out_p + (size_t)BB * SS * DD;           // [B,H,S,S]

  const size_t M = (size_t)BB * SS;                       // 8192
  uint8_t* ws = (uint8_t*)d_ws;
  __bf16* Qbf = (__bf16*)ws;                  ws += M * DD * sizeof(__bf16);
  __bf16* Kbf = (__bf16*)ws;                  ws += M * DD * sizeof(__bf16);
  __bf16* Vt  = (__bf16*)ws;                  ws += M * DD * sizeof(__bf16);
  float*  ctx = (float*)ws;                   ws += M * DD * sizeof(float);
  float*  opj = (float*)ws;

  const int gemm_blocks = (int)((M / 64) * (DD / 64));            // 2048
  const int score_blocks = BB * HH * (SS / 16) / 4;               // 2048
  const int ctx_blocks = BB * HH * (SS / 32) / 4;                 // 1024

  // QKV projections (bf16 out; V stored transposed per batch)
  gemm_xwt_kernel<true, false><<<gemm_blocks, 128, 0, stream>>>(query, Wq, bq, Qbf, (int)M, DD, DD);
  gemm_xwt_kernel<true, false><<<gemm_blocks, 128, 0, stream>>>(key_t, Wk, bk, Kbf, (int)M, DD, DD);
  gemm_xwt_kernel<true, true><<<gemm_blocks, 128, 0, stream>>>(value, Wv, bv, Vt, (int)M, DD, DD);

  // Raw scores -> attn region, then softmax in place (final attn output)
  attn_scores_kernel<<<score_blocks, 128, 0, stream>>>(Qbf, Kbf, attn_p);
  softmax_rows_kernel<<<BB * HH * SS, 256, 0, stream>>>(attn_p);

  // context = attn @ V, O projection (+bias), residual + LayerNorm
  attn_context_kernel<<<ctx_blocks, 128, 0, stream>>>(attn_p, Vt, ctx);
  gemm_xwt_kernel<false, false><<<gemm_blocks, 128, 0, stream>>>(ctx, Wo, bo, opj, (int)M, DD, DD);
  residual_ln_kernel<<<(int)M, 256, 0, stream>>>(opj, query, gamma, beta, out_p);
}
